// GroupedQueryAttention_39848706573773
// MI455X (gfx1250) — matrix operand order repaired and checked by reading, not compiled, not run on hardware
//
#include <hip/hip_runtime.h>
#include <hip/hip_bf16.h>

// ---------------------------------------------------------------------------
// CDNA5 (gfx1250) GQA attention: bf16 WMMA everywhere, fp32 accumulate.
// ---------------------------------------------------------------------------

typedef __attribute__((ext_vector_type(16))) __bf16 v16bf;
typedef __attribute__((ext_vector_type(8)))  float  v8f;

union Frag16 {
    v16bf v;
    uint4 q[2];
};

__device__ __forceinline__ v8f wmma_bf16(const Frag16& a, const Frag16& b, v8f c) {
    // D = A(16x32 bf16) * B(32x16 bf16) + C(16x16 f32)
    return __builtin_amdgcn_wmma_f32_16x16x32_bf16(
        /*neg_a=*/false, a.v, /*neg_b=*/false, b.v,
        /*c_mod=*/(short)0, c, /*reuse_a=*/false, /*reuse_b=*/false);
}

__device__ __forceinline__ float rowmax16(float v) {
#pragma unroll
    for (int off = 8; off; off >>= 1) v = fmaxf(v, __shfl_xor(v, off));
    return v;
}
__device__ __forceinline__ float rowsum16(float v) {
#pragma unroll
    for (int off = 8; off; off >>= 1) v += __shfl_xor(v, off);
    return v;
}

// ---------------------------------------------------------------------------
// fp32 -> bf16 conversion (grid-stride)
// ---------------------------------------------------------------------------
__global__ __launch_bounds__(256)
void cvt_f32_bf16(const float* __restrict__ src, __hip_bfloat16* __restrict__ dst,
                  size_t n) {
    size_t i = (size_t)blockIdx.x * blockDim.x + threadIdx.x;
    const size_t stride = (size_t)gridDim.x * blockDim.x;
    for (; i < n; i += stride) dst[i] = __float2bfloat16(src[i]);
}

// ---------------------------------------------------------------------------
// C[M,N] = out_scale * (A[M,K] * W[N,K]^T)   (torch Linear). A,W bf16.
// Block tile 128(M) x 128(N), K in steps of 32, double-buffered LDS.
// 8 waves as 4(M) x 2(N): each wave = 2 M-subtiles x 4 N-subtiles = 8 WMMAs
// per K-step fed by 12 ds_load_b128 (B frags reused across M-subtiles).
// Global loads for tile k+1 are held in registers across the compute phase.
// All problem dims are multiples of the tile sizes (4096/2048/512).
// ---------------------------------------------------------------------------
template <int OUT_F32>
__global__ __launch_bounds__(256)
void gemm_nt_wmma(const __hip_bfloat16* __restrict__ A,
                  const __hip_bfloat16* __restrict__ W,
                  void* __restrict__ Cout, int M, int N, int K, float out_scale) {
    __shared__ alignas(16) __hip_bfloat16 As[2][128 * 32];  // 16 KB
    __shared__ alignas(16) __hip_bfloat16 Ws[2][128 * 32];  // 16 KB
    const int tid = threadIdx.x, wid = tid >> 5, lane = tid & 31;
    const int ln = lane & 15, half = lane >> 4;
    const int wm = wid & 3, wn = wid >> 2;  // 4 x 2 wave grid
    const int m0 = blockIdx.x * 128, n0 = blockIdx.y * 128;

    // staging slice: 16 elems (32B) of A and of W per thread
    const int srow = tid >> 1, sko = (tid & 1) * 16;
    const __hip_bfloat16* aptr = A + (size_t)(m0 + srow) * K + sko;
    const __hip_bfloat16* wptr = W + (size_t)(n0 + srow) * K + sko;

    // prologue: stage tile 0
    {
        const uint4* asrc = (const uint4*)(aptr);
        uint4* adst = (uint4*)(&As[0][srow * 32 + sko]);
        adst[0] = asrc[0];
        adst[1] = asrc[1];
        const uint4* wsrc = (const uint4*)(wptr);
        uint4* wdst = (uint4*)(&Ws[0][srow * 32 + sko]);
        wdst[0] = wsrc[0];
        wdst[1] = wsrc[1];
    }

    v8f acc[2][4] = {};
    int buf = 0;
    for (int k0 = 0; k0 < K; k0 += 32, buf ^= 1) {
        __syncthreads();
        const bool has_next = (k0 + 32) < K;
        uint4 ta0{}, ta1{}, tw0{}, tw1{};
        if (has_next) {  // global loads issue here, land during the WMMAs below
            const uint4* asrc = (const uint4*)(aptr + k0 + 32);
            ta0 = asrc[0];
            ta1 = asrc[1];
            const uint4* wsrc = (const uint4*)(wptr + k0 + 32);
            tw0 = wsrc[0];
            tw1 = wsrc[1];
            __builtin_prefetch(aptr + k0 + 64, 0, 0);  // global_prefetch_b8
            __builtin_prefetch(wptr + k0 + 64, 0, 0);
        }

        // A fragments (16x32): lane halves hold K 0-7/16-23 vs 8-15/24-31
        Frag16 af[2];
#pragma unroll
        for (int ms = 0; ms < 2; ++ms) {
            const uint4* pa = (const uint4*)&As[buf][(wm * 32 + ms * 16 + ln) * 32];
            af[ms].q[0] = pa[half];
            af[ms].q[1] = pa[2 + half];
        }
#pragma unroll
        for (int n = 0; n < 4; ++n) {
            // B fragment (32x16): lane = column, halves hold K 0-7/16-23 vs 8-15/24-31.
            Frag16 bf_;
            const uint4* pb =
                (const uint4*)&Ws[buf][(wn * 64 + n * 16 + ln) * 32];
            bf_.q[0] = pb[half];
            bf_.q[1] = pb[2 + half];
#pragma unroll
            for (int ms = 0; ms < 2; ++ms)
                acc[ms][n] = wmma_bf16(af[ms], bf_, acc[ms][n]);
        }

        if (has_next) {  // commit next tile into the other buffer
            uint4* adst = (uint4*)(&As[buf ^ 1][srow * 32 + sko]);
            adst[0] = ta0;
            adst[1] = ta1;
            uint4* wdst = (uint4*)(&Ws[buf ^ 1][srow * 32 + sko]);
            wdst[0] = tw0;
            wdst[1] = tw1;
        }
    }

    // C/D layout: VGPR r -> row r + 8*half, lane%16 -> col
#pragma unroll
    for (int ms = 0; ms < 2; ++ms) {
#pragma unroll
        for (int n = 0; n < 4; ++n) {
#pragma unroll
            for (int r = 0; r < 8; ++r) {
                const int row = m0 + wm * 32 + ms * 16 + half * 8 + r;
                const size_t idx = (size_t)row * N + n0 + wn * 64 + n * 16 + ln;
                const float v = acc[ms][n][r] * out_scale;
                if (OUT_F32)
                    ((float*)Cout)[idx] = v;
                else
                    ((__hip_bfloat16*)Cout)[idx] = __float2bfloat16(v);
            }
        }
    }
}

// ---------------------------------------------------------------------------
// Flash attention (causal, GQA). Block = 8 waves = 128 query rows, one head.
// Each wave runs online softmax for its 16 rows; the block cooperatively
// stages 64-key K/V tiles (V transposed in LDS so PV B-fragments are
// contiguous 32B reads). The next chunk's K/V global loads are issued into
// registers while the current chunk's S/softmax/PV compute runs.
// Q is pre-scaled by 1/sqrt(hd) at projection time.
// ---------------------------------------------------------------------------
__global__ __launch_bounds__(256)
void attn_fmha(const __hip_bfloat16* __restrict__ Qp,
               const __hip_bfloat16* __restrict__ Kp,
               const __hip_bfloat16* __restrict__ Vp,
               __hip_bfloat16* __restrict__ Op, int T) {
    constexpr int HD = 128, CQ = 2048, CKV = 512;
    __shared__ alignas(16) __hip_bfloat16 Ks[64 * HD];     // 16 KB, [key][d]
    __shared__ alignas(16) __hip_bfloat16 Vt[HD * 64];     // 16 KB, [d][key]
    __shared__ alignas(16) __hip_bfloat16 Ps[8][16 * 64];  // 16 KB, per-wave P

    const int tid = threadIdx.x, wid = tid >> 5, lane = tid & 31;
    const int ln = lane & 15, half = lane >> 4;
    const int qt = blockIdx.x, h = blockIdx.y, b = blockIdx.z;
    const int kvh = h >> 2;  // g = H/KV = 4
    const int qb = qt * 128;
    const int qrow = qb + wid * 16 + ln;

    // Q fragments: this wave's 16 rows x 128, 4 chunks of K=32, kept in VGPRs.
    Frag16 qf[4];
    const __hip_bfloat16* qptr = Qp + ((size_t)b * T + qrow) * CQ + h * HD;
#pragma unroll
    for (int kk = 0; kk < 4; ++kk) {
        const uint4* pq = (const uint4*)(qptr + kk * 32);
        qf[kk].q[0] = pq[half];
        qf[kk].q[1] = pq[2 + half];
    }

    v8f o_acc[8] = {};
    float m_run[8], l_run[8];
#pragma unroll
    for (int r = 0; r < 8; ++r) { m_run[r] = -1.0e30f; l_run[r] = 0.0f; }

    // staging slice: one key row, 32 d-elems of K and of V per thread
    const int srow = tid >> 2, sd0 = (tid & 3) * 32;
    uint4 kr[4], vr[4];
    {
        const size_t goff = ((size_t)b * T + srow) * CKV + kvh * HD + sd0;
        const uint4* ks = (const uint4*)(Kp + goff);
        kr[0] = ks[0]; kr[1] = ks[1]; kr[2] = ks[2]; kr[3] = ks[3];
        const uint4* vs = (const uint4*)(Vp + goff);
        vr[0] = vs[0]; vr[1] = vs[1]; vr[2] = vs[2]; vr[3] = vs[3];
    }

    const int nchunks = qb / 64 + 2;  // uniform per block -> barriers are safe
    for (int c = 0; c < nchunks; ++c) {
        // ---- commit the register-held chunk into LDS ----
        {
            uint4* kdst = (uint4*)(&Ks[srow * HD + sd0]);
            kdst[0] = kr[0]; kdst[1] = kr[1]; kdst[2] = kr[2]; kdst[3] = kr[3];
            const __hip_bfloat16* te = (const __hip_bfloat16*)vr;
#pragma unroll
            for (int j = 0; j < 32; ++j) Vt[(sd0 + j) * 64 + srow] = te[j];
        }
        __syncthreads();

        // ---- prefetch next chunk into registers (overlaps compute) ----
        if (c + 1 < nchunks) {
            const size_t goff =
                ((size_t)b * T + (c + 1) * 64 + srow) * CKV + kvh * HD + sd0;
            const uint4* ks = (const uint4*)(Kp + goff);
            kr[0] = ks[0]; kr[1] = ks[1]; kr[2] = ks[2]; kr[3] = ks[3];
            const uint4* vs = (const uint4*)(Vp + goff);
            vr[0] = vs[0]; vr[1] = vs[1]; vr[2] = vs[2]; vr[3] = vs[3];
        }

        // ---- S = Q K^T (16 x 64), 16 WMMAs ----
        const int kb = c * 64;
        v8f S[4];
#pragma unroll
        for (int n = 0; n < 4; ++n) {
            v8f s = {};
#pragma unroll
            for (int kk = 0; kk < 4; ++kk) {
                Frag16 kf;  // B[k][n] = Key[n][k] -> contiguous Ks row read
                const uint4* pk =
                    (const uint4*)&Ks[(n * 16 + ln) * HD + kk * 32];
                kf.q[0] = pk[half];
                kf.q[1] = pk[2 + half];
                s = wmma_bf16(qf[kk], kf, s);
            }
            S[n] = s;
        }

        // ---- causal mask + online softmax (scale already folded into Q) ----
        float cmax[8];
#pragma unroll
        for (int r = 0; r < 8; ++r) cmax[r] = -1.0e30f;
#pragma unroll
        for (int n = 0; n < 4; ++n) {
            const int tk = kb + n * 16 + ln;
#pragma unroll
            for (int r = 0; r < 8; ++r) {
                const int tq = qb + wid * 16 + half * 8 + r;
                float sv = S[n][r];
                sv = (tk > tq) ? -1.0e30f : sv;
                S[n][r] = sv;
                cmax[r] = fmaxf(cmax[r], sv);
            }
        }
        float alpha[8], rsum[8];
#pragma unroll
        for (int r = 0; r < 8; ++r) {
            const float mnew = fmaxf(m_run[r], rowmax16(cmax[r]));
            alpha[r] = __expf(m_run[r] - mnew);
            m_run[r] = mnew;
            rsum[r] = 0.0f;
        }
#pragma unroll
        for (int n = 0; n < 4; ++n) {
#pragma unroll
            for (int r = 0; r < 8; ++r) {
                const float pexp = __expf(S[n][r] - m_run[r]);
                rsum[r] += pexp;
                Ps[wid][(half * 8 + r) * 64 + n * 16 + ln] = __float2bfloat16(pexp);
            }
        }
#pragma unroll
        for (int r = 0; r < 8; ++r) {
            l_run[r] = l_run[r] * alpha[r] + rowsum16(rsum[r]);
#pragma unroll
            for (int n = 0; n < 8; ++n) o_acc[n][r] *= alpha[r];
        }
        __syncthreads();  // Ps C-layout -> A-layout handoff

        // ---- O += P V (16 WMMAs) ----
        Frag16 pf[2];
        {
            const uint4* pp = (const uint4*)&Ps[wid][ln * 64];
#pragma unroll
            for (int kk = 0; kk < 2; ++kk) {
                pf[kk].q[0] = pp[kk * 4 + half];
                pf[kk].q[1] = pp[kk * 4 + 2 + half];
            }
        }
#pragma unroll
        for (int n = 0; n < 8; ++n) {
#pragma unroll
            for (int kk = 0; kk < 2; ++kk) {
                Frag16 vf;  // B[k][d] = V[k][d] -> contiguous Vt row read
                const uint4* pv =
                    (const uint4*)&Vt[(n * 16 + ln) * 64 + kk * 32];
                vf.q[0] = pv[half];
                vf.q[1] = pv[2 + half];
                o_acc[n] = wmma_bf16(pf[kk], vf, o_acc[n]);
            }
        }
        __syncthreads();  // protect Ks/Vt before next commit
    }

    // ---- epilogue: normalize, write bf16 attn output [B,T,C] ----
#pragma unroll
    for (int r = 0; r < 8; ++r) {
        const float inv_l = 1.0f / l_run[r];
        const int row = qb + wid * 16 + half * 8 + r;
        __hip_bfloat16* op = Op + ((size_t)b * T + row) * CQ + h * HD + ln;
#pragma unroll
        for (int n = 0; n < 8; ++n)
            op[n * 16] = __float2bfloat16(o_acc[n][r] * inv_l);
    }
}

// ---------------------------------------------------------------------------
// Host launcher
// ---------------------------------------------------------------------------
extern "C" void kernel_launch(void* const* d_in, const int* in_sizes, int n_in,
                              void* d_out, int out_size, void* d_ws, size_t ws_size,
                              hipStream_t stream) {
    (void)in_sizes; (void)n_in; (void)out_size; (void)ws_size;
    const int B = 2, T = 2048, C = 2048, H = 16, KVD = 512;
    const int M = B * T;  // 4096
    const float qscale = 0.08838834764831845f;  // 1/sqrt(128)

    const float* x  = (const float*)d_in[0];
    const float* Wq = (const float*)d_in[1];
    const float* Wk = (const float*)d_in[2];
    const float* Wv = (const float*)d_in[3];
    const float* Wo = (const float*)d_in[4];

    __hip_bfloat16* p = (__hip_bfloat16*)d_ws;
    __hip_bfloat16* xb  = p; p += (size_t)M * C;    // 16 MB
    __hip_bfloat16* wqb = p; p += (size_t)C * C;    //  8 MB
    __hip_bfloat16* wkb = p; p += (size_t)KVD * C;  //  2 MB
    __hip_bfloat16* wvb = p; p += (size_t)KVD * C;  //  2 MB
    __hip_bfloat16* wob = p; p += (size_t)C * C;    //  8 MB
    __hip_bfloat16* qb  = p; p += (size_t)M * C;    // 16 MB
    __hip_bfloat16* kb  = p; p += (size_t)M * KVD;  //  4 MB
    __hip_bfloat16* vb  = p; p += (size_t)M * KVD;  //  4 MB
    __hip_bfloat16* ob  = p; p += (size_t)M * C;    // 16 MB

    cvt_f32_bf16<<<2048, 256, 0, stream>>>(x,  xb,  (size_t)M * C);
    cvt_f32_bf16<<<1024, 256, 0, stream>>>(Wq, wqb, (size_t)C * C);
    cvt_f32_bf16<<<512,  256, 0, stream>>>(Wk, wkb, (size_t)KVD * C);
    cvt_f32_bf16<<<512,  256, 0, stream>>>(Wv, wvb, (size_t)KVD * C);
    cvt_f32_bf16<<<1024, 256, 0, stream>>>(Wo, wob, (size_t)C * C);

    // Projections: y = x @ W^T  (Q pre-scaled by 1/sqrt(hd))
    gemm_nt_wmma<0><<<dim3(M / 128, C / 128),   256, 0, stream>>>(xb, wqb, qb, M, C,   C, qscale);
    gemm_nt_wmma<0><<<dim3(M / 128, KVD / 128), 256, 0, stream>>>(xb, wkb, kb, M, KVD, C, 1.0f);
    gemm_nt_wmma<0><<<dim3(M / 128, KVD / 128), 256, 0, stream>>>(xb, wvb, vb, M, KVD, C, 1.0f);

    // Causal GQA flash attention
    attn_fmha<<<dim3(T / 128, H, B), 256, 0, stream>>>(qb, kb, vb, ob, T);

    // Output projection, fp32 out
    gemm_nt_wmma<1><<<dim3(M / 128, C / 128), 256, 0, stream>>>(ob, wob, d_out, M, C, C, 1.0f);
}
